// SSDModelWithAnchorsAndNMS_41910290874782
// MI455X (gfx1250) — matrix operand order repaired and checked by reading, not compiled, not run on hardware
//
#include <hip/hip_runtime.h>
#include <hip/hip_bf16.h>
#include <vector>
#include <cstddef>

// ---------------------------------------------------------------------------
// Types for CDNA5 WMMA
// ---------------------------------------------------------------------------
typedef __attribute__((ext_vector_type(16))) _Float16 v16h;
typedef __attribute__((ext_vector_type(8)))  _Float16 v8h;
typedef __attribute__((ext_vector_type(8)))  float    v8f;

#define TOPK 200
#define NCLS 21
#define NMAX 2766   // (19*19 + 10*10) * 6 anchors

__device__ __forceinline__ float relu6f(float x) {
    return fminf(fmaxf(x, 0.0f), 6.0f);
}

// ---------------------------------------------------------------------------
// Fold BN into weights (non-GEMM convs; unpadded [cout][cpk] layout)
// ---------------------------------------------------------------------------
__global__ void k_fold_bn(const float* __restrict__ w, const float* __restrict__ gamma,
                          const float* __restrict__ beta, const float* __restrict__ mean,
                          const float* __restrict__ var,
                          _Float16* __restrict__ w16, float* __restrict__ bias,
                          int cout, int cpk) {
    int i = blockIdx.x * blockDim.x + threadIdx.x;
    int n = cout * cpk;
    if (i < n) {
        int co = i / cpk;
        float s = gamma[co] * rsqrtf(var[co] + 1e-5f);
        w16[i] = (_Float16)(w[i] * s);
    }
    if (i < cout) {
        float s = gamma[i] * rsqrtf(var[i] + 1e-5f);
        bias[i] = beta[i] - mean[i] * s;
    }
}

// Fold BN into zero-padded GEMM weight tile [coutP][cinP]
__global__ void k_fold_bn_pad(const float* __restrict__ w, const float* __restrict__ gamma,
                              const float* __restrict__ beta, const float* __restrict__ mean,
                              const float* __restrict__ var,
                              _Float16* __restrict__ w16, float* __restrict__ bias,
                              int cout, int cin, int coutP, int cinP) {
    int i = blockIdx.x * blockDim.x + threadIdx.x;
    int n = coutP * cinP;
    if (i < n) {
        int co = i / cinP, k = i - co * cinP;
        float v = 0.0f;
        if (co < cout && k < cin) {
            float s = gamma[co] * rsqrtf(var[co] + 1e-5f);
            v = w[(long)co * cin + k] * s;
        }
        w16[i] = (_Float16)v;
    }
    if (i < cout) {
        float s = gamma[i] * rsqrtf(var[i] + 1e-5f);
        bias[i] = beta[i] - mean[i] * s;
    }
}

// Plain weight copy (head convs: no BN)
__global__ void k_copy_w(const float* __restrict__ w, const float* __restrict__ b,
                         _Float16* __restrict__ w16, float* __restrict__ bias,
                         int n, int cout) {
    int i = blockIdx.x * blockDim.x + threadIdx.x;
    if (i < n)    w16[i]  = (_Float16)w[i];
    if (i < cout) bias[i] = b[i];
}

// f32 NCHW -> f16 transposed + zero-padded Xt[B][HWP][cinP]
__global__ void k_cvt16t(const float* __restrict__ X, _Float16* __restrict__ Xt,
                         int B, int cin, int HW, int cinP, int HWP) {
    long i = (long)blockIdx.x * blockDim.x + threadIdx.x;
    long tot = (long)B * HWP * cinP;
    if (i >= tot) return;
    int k = (int)(i % cinP); long t = i / cinP;
    int p = (int)(t % HWP); int b = (int)(t / HWP);
    float v = 0.0f;
    if (k < cin && p < HW) v = X[((long)b * cin + k) * HW + p];
    Xt[i] = (_Float16)v;
}

// ---------------------------------------------------------------------------
// 1x1 conv as implicit GEMM on WMMA: C[co, hw] = sum_ci W[co,ci] * X[ci,hw]
// Each wave computes a 16(M) x 32(N) tile; 4 waves/block cover 64 rows of M.
// All hot-loop loads are unconditional, contiguous 16B/32B vector loads thanks
// to zero-padded weight tile [coutP][cinP] and transposed Xt[B][HWP][cinP].
// A fragment: per-lane K chunks {k0+hi*16 .. +7} and {k0+hi*16+8 .. +7}.
// B fragment: lanes 0-15 hold K=0..15, lanes 16-31 hold K=16..31, col=lane&15.
// ---------------------------------------------------------------------------
__global__ __launch_bounds__(128)
void k_gemm1x1_wmma(const _Float16* __restrict__ W16, const float* __restrict__ bias,
                    const _Float16* __restrict__ Xt, float* __restrict__ Y,
                    int cinP, int cout, int coutP, int HW, int HWP, int act) {
    const int l   = threadIdx.x & 31;   // lane in wave32
    const int wv  = threadIdx.x >> 5;   // wave id in block (0..3)
    const int lo  = l & 15;
    const int hi  = l >> 4;
    const int hw0 = blockIdx.x * 32;
    const int co0 = blockIdx.y * 64 + wv * 16;
    const int b   = blockIdx.z;
    if (co0 >= coutP) return;           // wave-uniform: EXEC stays all-ones below

    const _Float16* Arow = W16 + (long)(co0 + lo) * cinP;
    const _Float16* B0   = Xt + ((long)b * HWP + hw0 + lo) * cinP;
    const _Float16* B1   = B0 + (long)16 * cinP;

    v8f c0 = {}, c1 = {};
    for (int k0 = 0; k0 < cinP; k0 += 32) {
        __builtin_prefetch(Arow + k0 + 32, 0, 1);             // global_prefetch
        v8h a_lo = *(const v8h*)(Arow + k0 + hi * 16);
        v8h a_hi = *(const v8h*)(Arow + k0 + hi * 16 + 8);
        v16h a = __builtin_shufflevector(a_lo, a_hi,
                                         0, 1, 2, 3, 4, 5, 6, 7,
                                         8, 9, 10, 11, 12, 13, 14, 15);
        v16h bb0 = *(const v16h*)(B0 + k0 + hi * 16);
        v16h bb1 = *(const v16h*)(B1 + k0 + hi * 16);
        c0 = __builtin_amdgcn_wmma_f32_16x16x32_f16(false, a, false, bb0,
                                                    (short)0, c0, false, false);
        c1 = __builtin_amdgcn_wmma_f32_16x16x32_f16(false, a, false, bb1,
                                                    (short)0, c1, false, false);
    }

    float* Yb = Y + (long)b * cout * HW;
    const int col0 = hw0 + lo;
    const int col1 = hw0 + 16 + lo;
#pragma unroll
    for (int v = 0; v < 8; ++v) {
        int row = co0 + hi * 8 + v;     // C/D layout: VGPR v -> M = v (+8 for hi)
        if (row < cout) {
            float bv = bias[row];
            if (col0 < HW) {
                float val = c0[v] + bv;
                if (act) val = relu6f(val);
                Yb[(long)row * HW + col0] = val;
            }
            if (col1 < HW) {
                float val = c1[v] + bv;
                if (act) val = relu6f(val);
                Yb[(long)row * HW + col1] = val;
            }
        }
    }
}

// ---------------------------------------------------------------------------
// Dense 3x3 conv (stem conv), pad=1, NCHW, optional ReLU6
// ---------------------------------------------------------------------------
__global__ void k_conv3x3(const float* __restrict__ X, const _Float16* __restrict__ W16,
                          const float* __restrict__ bias, float* __restrict__ Y,
                          int B, int cin, int Hin, int Win,
                          int cout, int Hout, int Wout, int stride, int act) {
    long i = (long)blockIdx.x * blockDim.x + threadIdx.x;
    long tot = (long)B * cout * Hout * Wout;
    if (i >= tot) return;
    int wx = (int)(i % Wout); long t = i / Wout;
    int hy = (int)(t % Hout); t /= Hout;
    int co = (int)(t % cout); int b = (int)(t / cout);
    float acc = bias[co];
    const _Float16* wp = W16 + (long)co * cin * 9;
    for (int ci = 0; ci < cin; ++ci) {
        const float* xp = X + (((long)b * cin + ci) * Hin) * Win;
        for (int ky = 0; ky < 3; ++ky) {
            int yy = hy * stride + ky - 1;
            if (yy < 0 || yy >= Hin) continue;
            for (int kx = 0; kx < 3; ++kx) {
                int xx = wx * stride + kx - 1;
                if (xx < 0 || xx >= Win) continue;
                acc += (float)wp[ci * 9 + ky * 3 + kx] * xp[yy * Win + xx];
            }
        }
    }
    if (act) acc = relu6f(acc);
    Y[i] = acc;
}

// ---------------------------------------------------------------------------
// Depthwise 3x3 conv, pad=1, ReLU6 always
// ---------------------------------------------------------------------------
__global__ void k_dwconv3x3(const float* __restrict__ X, const _Float16* __restrict__ W16,
                            const float* __restrict__ bias, float* __restrict__ Y,
                            int B, int C, int Hin, int Win,
                            int Hout, int Wout, int stride) {
    long i = (long)blockIdx.x * blockDim.x + threadIdx.x;
    long tot = (long)B * C * Hout * Wout;
    if (i >= tot) return;
    int wx = (int)(i % Wout); long t = i / Wout;
    int hy = (int)(t % Hout); t /= Hout;
    int c  = (int)(t % C);    int b = (int)(t / C);
    float acc = bias[c];
    const float* xp = X + (((long)b * C + c) * Hin) * Win;
    const _Float16* wp = W16 + (long)c * 9;
#pragma unroll
    for (int ky = 0; ky < 3; ++ky) {
        int yy = hy * stride + ky - 1;
        if (yy < 0 || yy >= Hin) continue;
#pragma unroll
        for (int kx = 0; kx < 3; ++kx) {
            int xx = wx * stride + kx - 1;
            if (xx < 0 || xx >= Win) continue;
            acc += (float)wp[ky * 3 + kx] * xp[yy * Win + xx];
        }
    }
    Y[i] = relu6f(acc);
}

// Residual add (in place into y)
__global__ void k_add(float* __restrict__ y, const float* __restrict__ x, long n) {
    long i = (long)blockIdx.x * blockDim.x + threadIdx.x;
    if (i < n) y[i] += x[i];
}

// ---------------------------------------------------------------------------
// SSD head conv 3x3 s1 p1, writes directly in transposed prediction layout:
//  out[(b*Ntot + nbase + hw*(cout/div) + co/div)*div + co%div]
// ---------------------------------------------------------------------------
__global__ void k_head_conv(const float* __restrict__ X, const _Float16* __restrict__ W16,
                            const float* __restrict__ bias, float* __restrict__ out,
                            int B, int cin, int H, int W,
                            int cout, int Ntot, int nbase, int div) {
    long i = (long)blockIdx.x * blockDim.x + threadIdx.x;
    long tot = (long)B * cout * H * W;
    if (i >= tot) return;
    int wx = (int)(i % W); long t = i / W;
    int hy = (int)(t % H); t /= H;
    int co = (int)(t % cout); int b = (int)(t / cout);
    float acc = bias[co];
    const _Float16* wp = W16 + (long)co * cin * 9;
    for (int ci = 0; ci < cin; ++ci) {
        const float* xp = X + (((long)b * cin + ci) * H) * W;
        for (int ky = 0; ky < 3; ++ky) {
            int yy = hy + ky - 1;
            if (yy < 0 || yy >= H) continue;
            for (int kx = 0; kx < 3; ++kx) {
                int xx = wx + kx - 1;
                if (xx < 0 || xx >= W) continue;
                acc += (float)wp[ci * 9 + ky * 3 + kx] * xp[yy * W + xx];
            }
        }
    }
    int hw = hy * W + wx;
    int anchors = cout / div;
    int n = nbase + hw * anchors + co / div;
    int j = co % div;
    out[((long)b * Ntot + n) * div + j] = acc;
}

// scores = max over classes, labels = first argmax
__global__ void k_score_label(const float* __restrict__ cls, float* __restrict__ scores,
                              int* __restrict__ labels, int B, int N) {
    long i = (long)blockIdx.x * blockDim.x + threadIdx.x;
    if (i >= (long)B * N) return;
    const float* c = cls + i * NCLS;
    float best = c[0]; int lab = 0;
    for (int j = 1; j < NCLS; ++j) {
        float v = c[j];
        if (v > best) { best = v; lab = j; }
    }
    scores[i] = best;
    labels[i] = lab;
}

// ---------------------------------------------------------------------------
// Greedy NMS (TOP_K fixed iterations), one workgroup per batch image.
// ---------------------------------------------------------------------------
__global__ __launch_bounds__(256)
void k_nms(const float* __restrict__ boxes, const float* __restrict__ scores_in,
           const int* __restrict__ labels, float* __restrict__ out, int B, int N) {
    const int b = blockIdx.x;
    const int tid = threadIdx.x;
    const float* bx = boxes + (long)b * N * 4;
    const float* sc = scores_in + (long)b * N;
    const float NEG_INF = -__builtin_inff();

    __shared__ float s_scr[NMAX];
    __shared__ float s_area[NMAX];
    __shared__ float s_red[256];
    __shared__ int   s_idx[256];
    __shared__ int   s_keep[TOPK];
    __shared__ unsigned char s_val[TOPK];
    __shared__ float s_box[4];
    __shared__ int   s_i;

    for (int j = tid; j < N; j += 256) {
        s_scr[j] = sc[j];
        float x0 = bx[j * 4], y0 = bx[j * 4 + 1], x1 = bx[j * 4 + 2], y1 = bx[j * 4 + 3];
        s_area[j] = (x1 - x0) * (y1 - y0);
    }
    __syncthreads();

    for (int k = 0; k < TOPK; ++k) {
        float best = NEG_INF; int bi = 0x7fffffff;
        for (int j = tid; j < N; j += 256) {
            float v = s_scr[j];
            if (v > best || (v == best && j < bi)) { best = v; bi = j; }
        }
        s_red[tid] = best; s_idx[tid] = bi;
        __syncthreads();
        for (int off = 128; off > 0; off >>= 1) {
            if (tid < off) {
                float v = s_red[tid + off]; int j = s_idx[tid + off];
                if (v > s_red[tid] || (v == s_red[tid] && j < s_idx[tid])) {
                    s_red[tid] = v; s_idx[tid] = j;
                }
            }
            __syncthreads();
        }
        if (tid == 0) {
            int i = s_idx[0];
            int any = (s_red[0] > NEG_INF) ? 1 : 0;
            s_i = i;
            s_box[0] = bx[i * 4];     s_box[1] = bx[i * 4 + 1];
            s_box[2] = bx[i * 4 + 2]; s_box[3] = bx[i * 4 + 3];
            s_keep[k] = any ? i : -1;
            s_val[k]  = (unsigned char)any;
        }
        __syncthreads();
        const int i = s_i;
        const float ax0 = s_box[0], ay0 = s_box[1], ax1 = s_box[2], ay1 = s_box[3];
        const float areaI = s_area[i];
        for (int j = tid; j < N; j += 256) {
            float lx = fmaxf(ax0, bx[j * 4]);
            float ly = fmaxf(ay0, bx[j * 4 + 1]);
            float rx = fminf(ax1, bx[j * 4 + 2]);
            float ry = fminf(ay1, bx[j * 4 + 3]);
            float iw = fmaxf(rx - lx, 0.0f);
            float ih = fmaxf(ry - ly, 0.0f);
            float inter = iw * ih;
            float iou = inter / (areaI + s_area[j] - inter);
            if (iou > 0.5f || j == i) s_scr[j] = NEG_INF;
        }
        __syncthreads();
    }

    float* boxes_out  = out;
    float* labels_out = out + (long)B * TOPK * 4;
    float* scores_out = labels_out + (long)B * TOPK;
    float* valid_out  = scores_out + (long)B * TOPK;
    for (int k = tid; k < TOPK; k += 256) {
        int i = s_keep[k];
        int v = s_val[k];
        int ii = (i < 0) ? 0 : i;
#pragma unroll
        for (int j = 0; j < 4; ++j)
            boxes_out[((long)b * TOPK + k) * 4 + j] = v ? bx[ii * 4 + j] : 0.0f;
        labels_out[(long)b * TOPK + k] = v ? (float)labels[(long)b * N + ii] : -1.0f;
        scores_out[(long)b * TOPK + k] = v ? sc[ii] : 0.0f;
        valid_out[(long)b * TOPK + k]  = v ? 1.0f : 0.0f;
    }
}

// ---------------------------------------------------------------------------
// Host orchestration
// ---------------------------------------------------------------------------
struct ConvSpec { int cin, cout, k, s, pad, groups, act; };

static inline int cdiv(long a, long b) { return (int)((a + b - 1) / b); }

extern "C" void kernel_launch(void* const* d_in, const int* in_sizes, int n_in,
                              void* d_out, int out_size, void* d_ws, size_t ws_size,
                              hipStream_t stream) {
    (void)in_sizes; (void)n_in; (void)out_size; (void)ws_size;
    const int B = 8;
    const int NTOT = NMAX;

    // --- rebuild layer specs (mirror of build_layer_specs) -----------------
    std::vector<std::vector<ConvSpec>> layers;
    std::vector<int> resf;
    layers.push_back({ConvSpec{3, 32, 3, 2, 1, 1, 1}});
    resf.push_back(0);
    int cin = 32;
    const int CFG[7][4] = {{1,16,1,1},{6,24,2,2},{6,32,3,2},{6,64,4,2},
                           {6,96,3,1},{6,160,3,2},{6,320,1,1}};
    for (int c = 0; c < 7; ++c) {
        int t = CFG[c][0], ch = CFG[c][1], n = CFG[c][2], s = CFG[c][3];
        for (int i = 0; i < n; ++i) {
            int stride = (i == 0) ? s : 1;
            int hidden = cin * t;
            std::vector<ConvSpec> convs;
            if (t != 1) convs.push_back(ConvSpec{cin, hidden, 1, 1, 0, 1, 1});
            convs.push_back(ConvSpec{hidden, hidden, 3, stride, 1, hidden, 1});
            convs.push_back(ConvSpec{hidden, ch, 1, 1, 0, 1, 0});
            layers.push_back(convs);
            resf.push_back((stride == 1 && cin == ch) ? 1 : 0);
            cin = ch;
        }
    }
    layers.push_back({ConvSpec{cin, 1280, 1, 1, 0, 1, 1}});
    resf.push_back(0);
    // features at layer indices 13 (96ch, 19x19) and 17 (320ch, 10x10)

    // --- workspace carve-up -------------------------------------------------
    char* ws = (char*)d_ws;
    size_t off = 0;
    auto alloc = [&](size_t bytes) -> void* {
        off = (off + 255) & ~(size_t)255;
        void* p = ws + off;
        off += bytes;
        return p;
    };
    const long MAXE = (long)B * 96 * 150 * 150;  // largest activation (17.28M)
    float*    buf[3];
    for (int i = 0; i < 3; ++i) buf[i] = (float*)alloc((size_t)MAXE * 4);
    _Float16* x16  = (_Float16*)alloc((size_t)MAXE * 2);   // transposed padded acts
    _Float16* w16  = (_Float16*)alloc((size_t)1000000 * 2);
    float*    bias = (float*)alloc(2048 * 4);
    float*    locp = (float*)alloc((size_t)B * NTOT * 4 * 4);
    float*    clsp = (float*)alloc((size_t)B * NTOT * NCLS * 4);
    float*    scor = (float*)alloc((size_t)B * NTOT * 4);
    int*      labl = (int*)alloc((size_t)B * NTOT * 4);

    const int HEAD0 = 1 + 52 * 5;  // 52 backbone convs x 5 tensors, after x

    const float* cur = (const float*)d_in[0];
    int curBuf = -1, curC = 3, curH = 300, curW = 300;
    int pi = 1;

    for (size_t li = 0; li < layers.size(); ++li) {
        if (li == layers.size() - 1) break;  // final 1280 conv is unused by outputs
        const float* resIn = nullptr;
        int resBuf = -1;
        if (resf[li]) { resIn = cur; resBuf = curBuf; }

        for (const ConvSpec& sp : layers[li]) {
            const float* w     = (const float*)d_in[pi + 0];
            const float* gamma = (const float*)d_in[pi + 1];
            const float* beta  = (const float*)d_in[pi + 2];
            const float* mean  = (const float*)d_in[pi + 3];
            const float* var   = (const float*)d_in[pi + 4];
            pi += 5;

            int Hout = (curH + 2 * sp.pad - sp.k) / sp.s + 1;
            int Wout = (curW + 2 * sp.pad - sp.k) / sp.s + 1;

            int ob = 0;
            while (ob == curBuf || ob == resBuf) ++ob;
            float* outp = buf[ob];

            if (sp.k == 1 && sp.groups == 1) {
                // ---- WMMA implicit GEMM path ----
                int cinP  = (sp.cin  + 31) & ~31;
                int coutP = (sp.cout + 15) & ~15;
                int HW    = curH * curW;
                int HWP   = (HW + 31) & ~31;
                long wn = (long)coutP * cinP;
                k_fold_bn_pad<<<cdiv(wn, 256), 256, 0, stream>>>(
                    w, gamma, beta, mean, var, w16, bias,
                    sp.cout, sp.cin, coutP, cinP);
                long nT = (long)B * HWP * cinP;
                k_cvt16t<<<cdiv(nT, 256), 256, 0, stream>>>(
                    cur, x16, B, sp.cin, HW, cinP, HWP);
                dim3 grid(HWP / 32, cdiv(coutP, 64), B);
                k_gemm1x1_wmma<<<grid, 128, 0, stream>>>(
                    w16, bias, x16, outp, cinP, sp.cout, coutP, HW, HWP, sp.act);
            } else {
                int cpk = (sp.cin / sp.groups) * sp.k * sp.k;
                long wn = (long)sp.cout * cpk;
                k_fold_bn<<<cdiv(wn, 256), 256, 0, stream>>>(
                    w, gamma, beta, mean, var, w16, bias, sp.cout, cpk);
                long tot = (long)B * sp.cout * Hout * Wout;
                if (sp.groups == 1) {  // stem 3x3
                    k_conv3x3<<<cdiv(tot, 256), 256, 0, stream>>>(
                        cur, w16, bias, outp, B, sp.cin, curH, curW,
                        sp.cout, Hout, Wout, sp.s, sp.act);
                } else {               // depthwise 3x3
                    k_dwconv3x3<<<cdiv(tot, 256), 256, 0, stream>>>(
                        cur, w16, bias, outp, B, sp.cout, curH, curW,
                        Hout, Wout, sp.s);
                }
            }
            cur = outp; curBuf = ob; curC = sp.cout; curH = Hout; curW = Wout;
        }

        if (resf[li]) {
            long n = (long)B * curC * curH * curW;
            k_add<<<cdiv(n, 256), 256, 0, stream>>>(buf[curBuf], resIn, n);
        }

        if (li == 13 || li == 17) {
            int f = (li == 13) ? 0 : 1;
            int nbase = (f == 0) ? 0 : 361 * 6;
            {   // loc head
                const float* hw = (const float*)d_in[HEAD0 + 2 * f + 0];
                const float* hb = (const float*)d_in[HEAD0 + 2 * f + 1];
                int cout = 24;
                long wn = (long)cout * curC * 9;
                k_copy_w<<<cdiv(wn, 256), 256, 0, stream>>>(hw, hb, w16, bias, (int)wn, cout);
                long tot = (long)B * cout * curH * curW;
                k_head_conv<<<cdiv(tot, 256), 256, 0, stream>>>(
                    cur, w16, bias, locp, B, curC, curH, curW, cout, NTOT, nbase, 4);
            }
            {   // cls head
                const float* hw = (const float*)d_in[HEAD0 + 4 + 2 * f + 0];
                const float* hb = (const float*)d_in[HEAD0 + 4 + 2 * f + 1];
                int cout = 6 * NCLS;
                long wn = (long)cout * curC * 9;
                k_copy_w<<<cdiv(wn, 256), 256, 0, stream>>>(hw, hb, w16, bias, (int)wn, cout);
                long tot = (long)B * cout * curH * curW;
                k_head_conv<<<cdiv(tot, 256), 256, 0, stream>>>(
                    cur, w16, bias, clsp, B, curC, curH, curW, cout, NTOT, nbase, NCLS);
            }
        }
    }

    k_score_label<<<cdiv((long)B * NTOT, 256), 256, 0, stream>>>(clsp, scor, labl, B, NTOT);
    k_nms<<<B, 256, 0, stream>>>(locp, scor, labl, (float*)d_out, B, NTOT);
}
